// PrunedMultiheadAttention_39711267619433
// MI455X (gfx1250) — compile-verified, matrix operand order checked by reading, not run on hardware
//
#include <hip/hip_runtime.h>
#include <math.h>

typedef __attribute__((ext_vector_type(16))) _Float16     v16h;
typedef __attribute__((ext_vector_type(8)))  float        v8f;
typedef __attribute__((ext_vector_type(4)))  unsigned int v4u;
typedef __attribute__((ext_vector_type(8)))  int          v8i;
typedef __attribute__((ext_vector_type(4)))  int          v4i;

constexpr int B    = 8;
constexpr int E    = 256;
constexpr int L    = 4096;
constexpr int TOPK = 256;
constexpr int H    = 8;
constexpr int HD   = 32;   // E / H

// ---------------------------------------------------------------------------
// WMMA helpers: CDNA5 v_wmma_f32_16x16x32_f16, wave32 fragment layouts
// (CDNA5 ISA §7.12.2). lane = g*16 + r (g = lane>>4, r = lane&15):
//   halves 0..7  -> K = h     + 8*g
//   halves 8..15 -> K = 8 + h + 8*g
// A: row M = r ; B: column N = r. C/D: VGPR i -> (M = i + 8*g, N = r).
// ---------------------------------------------------------------------------
__device__ __forceinline__ v8f wmma_f16(v16h a, v16h b, v8f c) {
  return __builtin_amdgcn_wmma_f32_16x16x32_f16(
      /*neg_a=*/false, a, /*neg_b=*/false, b,
      /*c_mod=*/(short)0, c, /*reuse_a=*/false, /*reuse_b=*/false);
}

// Row-major [rows][ld] f16 matrix (global or LDS); rows r0..+15, K k0..+31.
__device__ __forceinline__ v16h frag_rowmajor(const _Float16* __restrict__ base,
                                              int ld, int r0, int k0, int lane) {
  const int g = (lane >> 4) & 1;
  const int r = lane & 15;
  const _Float16* p = base + (size_t)(r0 + r) * (size_t)ld + k0;
  v16h f;
#pragma unroll
  for (int h = 0; h < 16; ++h) {
    const int kk = (h < 8) ? (h + 8 * g) : (8 + h + 8 * g);
    f[h] = p[kk];
  }
  return f;
}

// B fragment from a K-major source: element (n,k) at base[k*ld + n].
__device__ __forceinline__ v16h frag_colmajor(const _Float16* __restrict__ base,
                                              int ld, int n0, int k0, int lane) {
  const int g = (lane >> 4) & 1;
  const int r = lane & 15;
  v16h f;
#pragma unroll
  for (int h = 0; h < 16; ++h) {
    const int kk = (h < 8) ? (h + 8 * g) : (8 + h + 8 * g);
    f[h] = base[(size_t)(k0 + kk) * (size_t)ld + (n0 + r)];
  }
  return f;
}

// ---------------------------------------------------------------------------
// CDNA5 async / tensor data movers
// ---------------------------------------------------------------------------
// Per-lane async gather into LDS (CDNA5 ISA): VDST holds the
// LDS byte offset, VADDR the 64-bit global address. Tracked with ASYNCcnt.
__device__ __forceinline__ void async_gather_b32(void* lds_ptr, const float* g) {
  const unsigned int lds_off = (unsigned int)(uintptr_t)lds_ptr;  // addrspace(3) offset
  asm volatile("global_load_async_to_lds_b32 %0, %1, off"
               :: "v"(lds_off), "v"(g) : "memory");
}
__device__ __forceinline__ void wait_asynccnt0() {
  asm volatile("s_wait_asynccnt 0x0" ::: "memory");
}

// TDM: one contiguous 1-D tile of 2-byte elements, LDS padding of
// 16 B (pad_amount=3) every 128 DWORDs (pad_interval=6) => 528 B row pitch
// for a 512 B source row. Descriptor per CDNA5 ISA.
// This toolchain's builtin is the 6-arg form (extra zero-filled group).
__device__ __forceinline__ void tdm_load_rows_padded(void* lds_ptr,
                                                     const void* gptr,
                                                     unsigned int n_elems) {
  const unsigned long long ga = (unsigned long long)(uintptr_t)gptr;
  const unsigned int lds_off = (unsigned int)(uintptr_t)lds_ptr;
  v4u g0;
  g0[0] = 0x1u;                                    // count=1, user descriptor
  g0[1] = lds_off;                                 // lds_addr
  g0[2] = (unsigned int)ga;                        // global_addr[31:0]
  g0[3] = ((unsigned int)(ga >> 32) & 0x01FFFFFFu) // global_addr[56:32]
          | 0x80000000u;                           // type=2 ("image")
  v8i g1;
  g1[0] = (int)0x07910000u;       // data_size=1(2B), pad_en, interval=6, amount=3
  g1[1] = (int)(n_elems << 16);   // tensor_dim0[15:0]
  g1[2] = (int)((n_elems >> 16) | (1u << 16));  // tensor_dim0[31:16], tensor_dim1=1
  g1[3] = (int)(n_elems << 16);   // tile_dim0 (16-bit)
  g1[4] = 1;                      // tile_dim1=1, tile_dim2=0
  g1[5] = (int)n_elems;           // tensor_dim0_stride[31:0]
  g1[6] = 0;
  g1[7] = 0;
  v4i z4 = {0, 0, 0, 0};
  v8i z8 = {0, 0, 0, 0, 0, 0, 0, 0};
  __builtin_amdgcn_tensor_load_to_lds(g0, g1, z4, z4, z8, 0);
}

// ---------------------------------------------------------------------------
// Stage 1: q [B,E,L] f32  ->  qf16 [B,L,E] f16  (tiled LDS transpose)
// ---------------------------------------------------------------------------
__global__ __launch_bounds__(1024) void transpose_to_f16(
    const float* __restrict__ src, _Float16* __restrict__ dst) {
  __shared__ float tile[32][33];
  const int b  = blockIdx.x;
  const int e0 = blockIdx.y * 32;
  const int l0 = blockIdx.z * 32;
  tile[threadIdx.y][threadIdx.x] =
      src[(size_t)b * E * L + (size_t)(e0 + threadIdx.y) * L + l0 + threadIdx.x];
  __syncthreads();
  dst[(size_t)b * L * E + (size_t)(l0 + threadIdx.y) * E + e0 + threadIdx.x] =
      (_Float16)tile[threadIdx.x][threadIdx.y];
}

// ---------------------------------------------------------------------------
// Stage 2: f32 -> f16 weights: w16 = [Wq | Wk | Wv | Wout], each [256][256]
// ---------------------------------------------------------------------------
__global__ __launch_bounds__(256) void convert_weights(
    const float* __restrict__ in_proj, const float* __restrict__ out_proj,
    _Float16* __restrict__ w16) {
  const int i = blockIdx.x * 256 + threadIdx.x;
  if (i < 3 * E * E)          w16[i] = (_Float16)in_proj[i];
  else if (i < 4 * E * E)     w16[i] = (_Float16)out_proj[i - 3 * E * E];
}

// ---------------------------------------------------------------------------
// Stage 3: min_d[b,l] = min_s sum_e |k[b,l,e] - q[b,rand[b,s],e]|
// block = 1024 = (32 l) x (32 s); tiles staged by async-to-LDS gathers
// (the fills are transposing gathers: global stride L -> LDS row-major).
// ---------------------------------------------------------------------------
__global__ __launch_bounds__(1024) void l1_min_dist(
    const float* __restrict__ keys, const float* __restrict__ query,
    const int* __restrict__ rind, float* __restrict__ min_d) {
  constexpr int EC = 64;
  __shared__ float kt[32][EC + 1];
  __shared__ float qt[32][EC + 1];
  __shared__ int   sidx[32];
  __shared__ float redm[32][33];

  const int b   = blockIdx.x;
  const int l0  = blockIdx.y * 32;
  const int tid = threadIdx.x;
  const int l   = tid & 31;
  const int s   = tid >> 5;

  float best = 3.0e38f;
  for (int s0 = 0; s0 < TOPK; s0 += 32) {
    if (tid < 32) sidx[tid] = rind[b * TOPK + s0 + tid];
    float acc = 0.0f;
    for (int e0 = 0; e0 < E; e0 += EC) {
      __syncthreads();   // previous compute done; sidx visible
#pragma unroll
      for (int i = tid; i < 32 * EC; i += 1024) {
        const int e = i >> 5, ll = i & 31;
        async_gather_b32(&kt[ll][e],
            &keys[(size_t)b * E * L + (size_t)(e0 + e) * L + (l0 + ll)]);
      }
#pragma unroll
      for (int i = tid; i < 32 * EC; i += 1024) {
        const int e = i >> 5, ss = i & 31;
        async_gather_b32(&qt[ss][e],
            &query[(size_t)b * E * L + (size_t)(e0 + e) * L + sidx[ss]]);
      }
      wait_asynccnt0();
      __syncthreads();
#pragma unroll 8
      for (int e = 0; e < EC; ++e) acc += fabsf(kt[l][e] - qt[s][e]);
    }
    best = fminf(best, acc);
    __syncthreads();
  }
  redm[s][l] = best;
  __syncthreads();
  if (tid < 32) {
    float m = redm[0][tid];
#pragma unroll
    for (int j = 1; j < 32; ++j) m = fminf(m, redm[j][tid]);
    min_d[(size_t)b * L + l0 + tid] = m;
  }
}

// ---------------------------------------------------------------------------
// Stage 4: per-batch sorted top-256 smallest via rank-by-counting in LDS.
// ---------------------------------------------------------------------------
__global__ __launch_bounds__(1024) void topk_select(
    const float* __restrict__ min_d, int* __restrict__ idx) {
  __shared__ float d[L];
  const int b = blockIdx.x, tid = threadIdx.x;
  for (int i = tid; i < L; i += 1024) d[i] = min_d[(size_t)b * L + i];
  __syncthreads();
  for (int i = tid; i < L; i += 1024) {
    const float di = d[i];
    int rank = 0;
    for (int j = 0; j < L; ++j) {
      const float dj = d[j];
      rank += (dj < di) | ((dj == di) & (j < i));
    }
    if (rank < TOPK) idx[b * TOPK + rank] = i;
  }
}

// ---------------------------------------------------------------------------
// Stage 5: gather selected k/v rows into f16 [B*TOPK][E]
// ---------------------------------------------------------------------------
__global__ __launch_bounds__(256) void gather_kv(
    const float* __restrict__ keys, const float* __restrict__ values,
    const int* __restrict__ idx, _Float16* __restrict__ sk,
    _Float16* __restrict__ sv) {
  const int bt = blockIdx.x;
  const int b  = bt >> 8;
  const int ls = idx[bt];
  const int e  = threadIdx.x;
  const size_t gsrc = (size_t)b * E * L + (size_t)e * L + ls;
  sk[(size_t)bt * E + e] = (_Float16)keys[gsrc];
  sv[(size_t)bt * E + e] = (_Float16)values[gsrc];
}

// ---------------------------------------------------------------------------
// Stage 6: WMMA GEMM: C[M,N] = (A[M,K] @ B[N,K]^T + bias) * scale
// block = 256 thr = 8 waves, tile 64x32 (wave -> one 16x16 C tile).
// The block's 16 KB weight tile (32 rows x K=256 f16, contiguous) is staged
// into LDS by ONE TDM transfer with 16 B/row padding (528 B pitch => the
// ds_load_b128 fragment reads advance 4 banks per row: conflict-free).
// Cf16 != null: f16 row-major [M][N]; else f32 transposed into [B,E,L].
// ---------------------------------------------------------------------------
__global__ __launch_bounds__(256) void gemm_bias_wmma(
    const _Float16* __restrict__ A, const _Float16* __restrict__ Bw,
    const float* __restrict__ bias, _Float16* __restrict__ Cf16,
    float* __restrict__ Cf32_tr, int M, int N, int K, float scale) {
  constexpr int BP = 264;                 // LDS pitch in halves (528 B)
  __shared__ _Float16 btile[32 * BP];     // 16.5 KB

  const int lane = threadIdx.x & 31;
  const int w    = threadIdx.x >> 5;
  const int m0   = blockIdx.x * 64 + (w >> 1) * 16;
  const int nloc = (w & 1) * 16;
  const int n0   = blockIdx.y * 32 + nloc;

  if (threadIdx.x < 32) {                 // wave 0 issues the TDM DMA
    tdm_load_rows_padded(btile, Bw + (size_t)blockIdx.y * 32 * K,
                         (unsigned int)(32 * K));
    __builtin_amdgcn_s_wait_tensorcnt(0);
  }
  __syncthreads();

  v8f c = {};
  for (int k0 = 0; k0 < K; k0 += 32) {
    v16h a = frag_rowmajor(A, K, m0, k0, lane);
    v16h b = frag_rowmajor(btile, BP, nloc, k0, lane);   // from LDS
    c = wmma_f16(a, b, c);
  }
  const int g = (lane >> 4) & 1, r = lane & 15;
  const int n = n0 + r;
  const float bn = bias ? bias[n] : 0.0f;
#pragma unroll
  for (int i = 0; i < 8; ++i) {
    const int   m = m0 + i + 8 * g;
    const float v = (c[i] + bn) * scale;
    if (Cf16) {
      Cf16[(size_t)m * N + n] = (_Float16)v;
    } else {
      const int bb = m >> 12;          // m = b*L + l, L = 4096
      const int ll = m & (L - 1);
      Cf32_tr[(size_t)bb * E * L + (size_t)n * L + ll] = v;
    }
  }
}

// ---------------------------------------------------------------------------
// Stage 7: fused attention per (b, h, 32-query tile):
//   scores(32x256, K=hd=32) -> f32 softmax in LDS -> o = attn @ vh (K=256)
// ---------------------------------------------------------------------------
__global__ __launch_bounds__(256) void attn_wmma_kernel(
    const _Float16* __restrict__ qh, const _Float16* __restrict__ kh,
    const _Float16* __restrict__ vh, _Float16* __restrict__ o16) {
  constexpr int SP = 264;                  // padded row (bank-conflict free)
  __shared__ float    sc[32][SP];          // 33 KB
  __shared__ _Float16 at[32][SP];          // 16.5 KB
  __shared__ float    red[32][8];

  const int bh = blockIdx.x;
  const int b  = bh >> 3, h = bh & 7;
  const int q0 = blockIdx.y * 32;
  const int tid  = threadIdx.x;
  const int lane = tid & 31, w = tid >> 5;
  const int g = (lane >> 4) & 1, r = lane & 15;

  // ---- scores: 2 m-tiles x 16 n-tiles; wave w -> mt = w&1, nt = (w>>1)*4+j
  {
    const int mt = w & 1;
    v16h a = frag_rowmajor(qh, E, b * L + q0 + mt * 16, h * HD, lane);
#pragma unroll
    for (int j = 0; j < 4; ++j) {
      const int nt = (w >> 1) * 4 + j;
      v16h kb = frag_rowmajor(kh, E, b * TOPK + nt * 16, h * HD, lane);
      v8f c = {};
      c = wmma_f16(a, kb, c);
#pragma unroll
      for (int i = 0; i < 8; ++i)
        sc[mt * 16 + i + 8 * g][nt * 16 + r] = c[i];
    }
  }
  __syncthreads();

  // ---- softmax over 256 columns: 8 threads per row (32 rows)
  {
    const int row = tid & 31, ch = tid >> 5;
    float mx = -3.0e38f;
#pragma unroll
    for (int e = 0; e < 32; ++e) mx = fmaxf(mx, sc[row][ch * 32 + e]);
    red[row][ch] = mx;
    __syncthreads();
    mx = red[row][0];
#pragma unroll
    for (int j = 1; j < 8; ++j) mx = fmaxf(mx, red[row][j]);
    __syncthreads();
    float sm = 0.0f;
#pragma unroll
    for (int e = 0; e < 32; ++e) {
      const float v = __expf(sc[row][ch * 32 + e] - mx);
      sc[row][ch * 32 + e] = v;
      sm += v;
    }
    red[row][ch] = sm;
    __syncthreads();
    float tot = 0.0f;
#pragma unroll
    for (int j = 0; j < 8; ++j) tot += red[row][j];
    const float inv = 1.0f / tot;
#pragma unroll
    for (int e = 0; e < 32; ++e)
      at[row][ch * 32 + e] = (_Float16)(sc[row][ch * 32 + e] * inv);
  }
  __syncthreads();

  // ---- o = attn(32x256) @ vh_head(256x32): 4 C tiles on waves 0..3
  if (w < 4) {
    const int mt = w & 1, nt = w >> 1;
    v8f c = {};
    for (int k0 = 0; k0 < TOPK; k0 += 32) {
      v16h a;
#pragma unroll
      for (int hh = 0; hh < 16; ++hh) {
        const int kk = (hh < 8) ? (hh + 8 * g) : (8 + hh + 8 * g);
        a[hh] = at[mt * 16 + r][k0 + kk];
      }
      v16h vb = frag_colmajor(vh + (size_t)(b * TOPK) * E + h * HD,
                              E, nt * 16, k0, lane);
      c = wmma_f16(a, vb, c);
    }
#pragma unroll
    for (int i = 0; i < 8; ++i) {
      const int q = q0 + mt * 16 + i + 8 * g;
      o16[(size_t)(b * L + q) * E + h * HD + nt * 16 + r] = (_Float16)c[i];
    }
  }
}

// ---------------------------------------------------------------------------
extern "C" void kernel_launch(void* const* d_in, const int* in_sizes, int n_in,
                              void* d_out, int out_size, void* d_ws, size_t ws_size,
                              hipStream_t stream) {
  (void)in_sizes; (void)n_in; (void)out_size; (void)ws_size;

  const float* query  = (const float*)d_in[0];
  const float* keys   = (const float*)d_in[1];
  const float* values = (const float*)d_in[2];
  const int*   rind   = (const int*)d_in[3];
  const float* in_w   = (const float*)d_in[4];
  const float* in_b   = (const float*)d_in[5];
  const float* out_w  = (const float*)d_in[6];
  const float* out_b  = (const float*)d_in[7];
  float*       out    = (float*)d_out;

  char* ws = (char*)d_ws;
  size_t off = 0;
  auto take = [&](size_t bytes) -> char* {
    char* p = ws + off;
    off = (off + bytes + 255) & ~(size_t)255;
    return p;
  };
  float*     mind = (float*)    take((size_t)B * L * 4);
  int*       idx  = (int*)      take((size_t)B * TOPK * 4);
  _Float16*  w16  = (_Float16*) take((size_t)4 * E * E * 2);   // Wq|Wk|Wv|Wout
  _Float16*  q16  = (_Float16*) take((size_t)B * L * E * 2);
  _Float16*  sk16 = (_Float16*) take((size_t)B * TOPK * E * 2);
  _Float16*  sv16 = (_Float16*) take((size_t)B * TOPK * E * 2);
  _Float16*  qh16 = (_Float16*) take((size_t)B * L * E * 2);
  _Float16*  kh16 = (_Float16*) take((size_t)B * TOPK * E * 2);
  _Float16*  vh16 = (_Float16*) take((size_t)B * TOPK * E * 2);
  _Float16*  o16  = (_Float16*) take((size_t)B * L * E * 2);

  // q -> [B,L,E] f16
  transpose_to_f16<<<dim3(B, E / 32, L / 32), dim3(32, 32), 0, stream>>>(query, q16);
  // weights -> f16
  convert_weights<<<dim3((4 * E * E) / 256), dim3(256), 0, stream>>>(in_w, out_w, w16);
  // L1 pruning distances
  l1_min_dist<<<dim3(B, L / 32), dim3(1024), 0, stream>>>(keys, query, rind, mind);
  // sorted top-k indices
  topk_select<<<dim3(B), dim3(1024), 0, stream>>>(mind, idx);
  // gather selected k/v rows (f16)
  gather_kv<<<dim3(B * TOPK), dim3(256), 0, stream>>>(keys, values, idx, sk16, sv16);

  const float inv_sqrt_hd = 0.17677669529663689f;  // 1/sqrt(32)
  // qh = (q @ Wq^T + bq) * scale        M = B*L = 32768
  gemm_bias_wmma<<<dim3((B * L) / 64, E / 32), dim3(256), 0, stream>>>(
      q16, w16 + 0 * E * E, in_b + 0 * E, qh16, nullptr, B * L, E, E, inv_sqrt_hd);
  // kh = sk @ Wk^T + bk                 M = B*TOPK = 2048
  gemm_bias_wmma<<<dim3((B * TOPK) / 64, E / 32), dim3(256), 0, stream>>>(
      sk16, w16 + 1 * E * E, in_b + 1 * E, kh16, nullptr, B * TOPK, E, E, 1.0f);
  // vh = sv @ Wv^T + bv
  gemm_bias_wmma<<<dim3((B * TOPK) / 64, E / 32), dim3(256), 0, stream>>>(
      sv16, w16 + 2 * E * E, in_b + 2 * E, vh16, nullptr, B * TOPK, E, E, 1.0f);
  // fused attention -> o16 [B*L][E]
  attn_wmma_kernel<<<dim3(B * H, L / 32), dim3(256), 0, stream>>>(qh16, kh16, vh16, o16);
  // out = o @ Wout^T + bout, written transposed into [B,E,L] f32
  gemm_bias_wmma<<<dim3((B * L) / 64, E / 32), dim3(256), 0, stream>>>(
      o16, w16 + 3 * E * E, out_b, nullptr, out, B * L, E, E, 1.0f);
}
